// unit_gtcn_7_32014686224462
// MI455X (gfx1250) — matrix operand order repaired and checked by reading, not compiled, not run on hardware
//
#include <hip/hip_runtime.h>
#include <hip/hip_bf16.h>
#include <math.h>

// Problem constants (from setup_inputs)
#define NB   16      // batch
#define CC   256     // channels
#define TT   64      // time
#define VV   25      // vertices
#define ICH  64      // inter channels
#define TVC  (TT*VV)     // 1600
#define ICT  (ICH*TT)    // 4096
#define K9   (CC*9)      // 2304
#define MDIM (CC*TT)     // 16384
#define BN_EPS 1e-5f

typedef __attribute__((ext_vector_type(16))) __bf16 v16bf;
typedef __attribute__((ext_vector_type(8)))  __bf16 bf16x8;
typedef __attribute__((ext_vector_type(8)))  float  v8f;

// ---------------------------------------------------------------------
// Fragment loaders (CDNA5 ISA 7.12.2, wave32).
// Operand buffers are laid out so each lane's 16 values are two
// contiguous 16-byte chunks -> global_load_b128 pairs.
// A (16x32): lane half hi: K = hi*8 + {0..7} and 16 + hi*8 + {0..7}
// B (32x16): lane half hi: K = hi*8 + {0..7} and 16 + hi*8 + {0..7}
// ---------------------------------------------------------------------
__device__ __forceinline__ v16bf frag_cat(bf16x8 lo, bf16x8 hi8) {
    return __builtin_shufflevector(lo, hi8, 0,1,2,3,4,5,6,7,8,9,10,11,12,13,14,15);
}
__device__ __forceinline__ v16bf load_Afrag(const __bf16* row, int k0, int hi) {
    bf16x8 c0 = *(const bf16x8*)(row + k0 + hi * 8);
    bf16x8 c1 = *(const bf16x8*)(row + k0 + 16 + hi * 8);
    return frag_cat(c0, c1);
}
__device__ __forceinline__ v16bf load_Bfrag(const __bf16* row, int k0, int hi) {
    bf16x8 c0 = *(const bf16x8*)(row + k0 + hi * 8);
    bf16x8 c1 = *(const bf16x8*)(row + k0 + 16 + hi * 8);
    return frag_cat(c0, c1);
}
#define WMMA_BF16(A, B, C) \
    __builtin_amdgcn_wmma_f32_16x16x32_bf16(false, (A), false, (B), (short)0, (C), false, false)

// =====================================================================
// Preprocessing kernels (one-time per launch)
// =====================================================================
// xT[n][tv][c] = bf16(x[n][c][tv])
__global__ void transpose_x_bf16(const float* __restrict__ x, __bf16* __restrict__ xT) {
    size_t idx = (size_t)blockIdx.x * blockDim.x + threadIdx.x;
    size_t total = (size_t)NB * TVC * CC;
    if (idx >= total) return;
    size_t n = idx / ((size_t)TVC * CC);
    size_t r = idx - n * (size_t)TVC * CC;
    size_t tv = r / CC, c = r - tv * CC;
    xT[idx] = (__bf16)x[(n * CC + c) * TVC + tv];
}
// straight f32 -> bf16
__global__ void cvt_bf16(const float* __restrict__ in, __bf16* __restrict__ out, size_t total) {
    size_t idx = (size_t)blockIdx.x * blockDim.x + threadIdx.x;
    if (idx < total) out[idx] = (__bf16)in[idx];
}
// [G][C][9] -> [G][9][C]  (kt-major conv9 weights)
__global__ void cvt_w9_bf16(const float* __restrict__ in, __bf16* __restrict__ out, size_t total) {
    size_t idx = (size_t)blockIdx.x * blockDim.x + threadIdx.x;
    if (idx >= total) return;
    size_t g = idx / (9 * CC);
    size_t r = idx - g * (9 * CC);
    size_t kt = r / CC, c = r - kt * CC;
    out[idx] = (__bf16)in[(g * CC + c) * 9 + kt];
}
__global__ void zero_bf16(__bf16* __restrict__ p, size_t total) {
    size_t idx = (size_t)blockIdx.x * blockDim.x + threadIdx.x;
    if (idx < total) p[idx] = (__bf16)0.0f;
}

// =====================================================================
// Static graph: A_all[s] = (4A^2-A-2I) + softmax_col((8A^4-4A^2-4A+I)/V) + PA
// =====================================================================
__global__ void build_Aall(const float* __restrict__ A, const float* __restrict__ PA,
                           float* __restrict__ Aall) {
    int s = blockIdx.x;
    int w = threadIdx.x, v = threadIdx.y;
    __shared__ float Mc[VV][VV];
    float a  = A[(s * VV + v) * VV + w];
    float a2 = a * a, a4 = a2 * a2;
    float eye = (v == w) ? 1.0f : 0.0f;
    float ch4 = (8.0f * a4 - 4.0f * a2 - 4.0f * a + eye) * (1.0f / (float)VV);
    Mc[v][w] = ch4;
    __syncthreads();
    float mx = -1e30f;
    for (int r = 0; r < VV; ++r) mx = fmaxf(mx, Mc[r][w]);
    float sm = 0.0f;
    for (int r = 0; r < VV; ++r) sm += __expf(Mc[r][w] - mx);
    float soft = __expf(ch4 - mx) / sm;
    float ch3 = 4.0f * a2 - a - 2.0f * eye;
    Aall[(s * VV + v) * VV + w] = ch3 + soft + PA[(s * VV + v) * VV + w];
}

__global__ void init_Ai(const float* __restrict__ Asrc, float* __restrict__ Ai) {
    int idx = blockIdx.x * blockDim.x + threadIdx.x;
    if (idx >= NB * VV * VV) return;
    Ai[idx] = Asrc[idx % (VV * VV)];
}

// =====================================================================
// Conv output stores.
// mode 0: fT[n][v(pad32)][o*T + t]   (feeds attention scores)
// mode 1: u [n][o*T + t][v(pad32)]   (feeds apply_adj A side)
// =====================================================================
__device__ __forceinline__ void store_conv_tile(v8f acc, int n, int oT, int hi, int col,
                                                const float* bias, __bf16* out,
                                                int O, int mode) {
    int t = col / VV, v = col - t * VV;
#pragma unroll
    for (int r = 0; r < 8; ++r) {
        int o = oT * 16 + (hi << 3) + r;
        float val = acc[r] + (bias ? bias[o] : 0.0f);
        size_t m = (size_t)o * TT + t;
        if (mode == 0)
            out[((size_t)n * 32 + v) * ((size_t)O * TT) + m] = (__bf16)val;
        else
            out[((size_t)n * O * TT + m) * 32 + v] = (__bf16)val;
    }
}

// =====================================================================
// conv1x1: out[n,o,tv] = sum_c W[o,c]*x[n,c,tv] (+bias).  W row stride wso,
// K=c contiguous (bf16).  Two 16-col tiles per wave share the A fragment.
// grid (TV/32, O/16, N), block 32
// =====================================================================
__global__ void __launch_bounds__(32)
conv1x1_wmma(const __bf16* __restrict__ xT, const __bf16* __restrict__ w,
             const float* __restrict__ bias, __bf16* __restrict__ out,
             int O, int wso, int mode) {
    int lane = threadIdx.x & 31;
    int hi = lane >> 4, lm = lane & 15;
    int n = blockIdx.z, oT = blockIdx.y, cT = blockIdx.x;
    const __bf16* xn = xT + (size_t)n * TVC * CC;
    int col0 = cT * 32 + lm, col1 = col0 + 16;
    const __bf16* wr  = w + (size_t)(oT * 16 + lm) * wso;
    const __bf16* xr0 = xn + (size_t)col0 * CC;
    const __bf16* xr1 = xn + (size_t)col1 * CC;
    v8f acc0 = {}, acc1 = {};
    for (int k0 = 0; k0 < CC; k0 += 32) {
        v16bf a  = load_Afrag(wr, k0, hi);
        v16bf b0 = load_Bfrag(xr0, k0, hi);
        v16bf b1 = load_Bfrag(xr1, k0, hi);
        acc0 = WMMA_BF16(a, b0, acc0);
        acc1 = WMMA_BF16(a, b1, acc1);
    }
    store_conv_tile(acc0, n, oT, hi, col0, bias, out, O, mode);
    store_conv_tile(acc1, n, oT, hi, col1, bias, out, O, mode);
}

// =====================================================================
// conv9: out[n,o,t,v] = sum_{c,kt} W[o][kt*256+c] * x[n,c,t+kt-4,v] (+bias)
// K = kt*256 + c (2304); kt uniform within each 32-step, c contiguous.
// grid (TV/32, O/16, N), block 32
// =====================================================================
__global__ void __launch_bounds__(32)
conv9_wmma(const __bf16* __restrict__ xT, const __bf16* __restrict__ w,
           const float* __restrict__ bias, __bf16* __restrict__ out,
           int O, int mode) {
    int lane = threadIdx.x & 31;
    int hi = lane >> 4, lm = lane & 15;
    int n = blockIdx.z, oT = blockIdx.y, cT = blockIdx.x;
    const __bf16* xn = xT + (size_t)n * TVC * CC;
    int col0 = cT * 32 + lm, col1 = col0 + 16;
    int t0 = col0 / VV, v0 = col0 - t0 * VV;
    int t1 = col1 / VV, v1 = col1 - t1 * VV;
    const __bf16* wr = w + (size_t)(oT * 16 + lm) * K9;
    v8f acc0 = {}, acc1 = {};
    v16bf zf = {};
    for (int kk0 = 0; kk0 < K9; kk0 += 32) {
        int kt = kk0 >> 8;               // 256 | step boundaries
        int c0 = kk0 - (kt << 8);
        v16bf a = load_Afrag(wr, kk0, hi);
        int tp0 = t0 + kt - 4;
        int tp1 = t1 + kt - 4;
        v16bf b0 = (tp0 >= 0 && tp0 < TT)
                 ? load_Bfrag(xn + ((size_t)tp0 * VV + v0) * CC, c0, hi) : zf;
        v16bf b1 = (tp1 >= 0 && tp1 < TT)
                 ? load_Bfrag(xn + ((size_t)tp1 * VV + v1) * CC, c0, hi) : zf;
        acc0 = WMMA_BF16(a, b0, acc0);
        acc1 = WMMA_BF16(a, b1, acc1);
    }
    store_conv_tile(acc0, n, oT, hi, col0, bias, out, O, mode);
    store_conv_tile(acc1, n, oT, hi, col1, bias, out, O, mode);
}

// =====================================================================
// scores: S[n,v,w] = sum_k fT1[n][v][k] * fT2[n][w][k], K = IC*T = 4096
// both operands row-contiguous in the v-major layout. grid (2,2,N)
// =====================================================================
__global__ void __launch_bounds__(32)
attn_scores_wmma(const __bf16* __restrict__ fT1, const __bf16* __restrict__ fT2,
                 float* __restrict__ S) {
    int lane = threadIdx.x & 31;
    int hi = lane >> 4, lm = lane & 15;
    int n = blockIdx.z, vT = blockIdx.y, wT = blockIdx.x;
    int vrow = vT * 16 + lm;
    int wcol = wT * 16 + lm;
    const __bf16* r1 = fT1 + ((size_t)n * 32 + vrow) * ICT;
    const __bf16* r2 = fT2 + ((size_t)n * 32 + wcol) * ICT;
    v8f acc = {};
    for (int k0 = 0; k0 < ICT; k0 += 32) {
        v16bf a = load_Afrag(r1, k0, hi);
        v16bf b = load_Bfrag(r2, k0, hi);
        acc = WMMA_BF16(a, b, acc);
    }
#pragma unroll
    for (int r = 0; r < 8; ++r) {
        int m = vT * 16 + (hi << 3) + r;
        if (m < VV && wcol < VV)
            S[((size_t)n * VV + m) * VV + wcol] = acc[r];
    }
}

// softmax over v (axis -2) of scaled scores, added into f32 Ai accumulator
__global__ void softmax_col_add(const float* __restrict__ S, float* __restrict__ Ai,
                                float scale) {
    int n = blockIdx.x / VV;
    int w = blockIdx.x - n * VV;
    int v = threadIdx.x;
    __shared__ float buf[32];
    __shared__ float red[2];
    float val = (v < VV) ? S[((size_t)n * VV + v) * VV + w] * scale : -1e30f;
    buf[v] = val;
    __syncthreads();
    if (v == 0) {
        float m = -1e30f;
        for (int r = 0; r < VV; ++r) m = fmaxf(m, buf[r]);
        float s = 0.0f;
        for (int r = 0; r < VV; ++r) s += __expf(buf[r] - m);
        red[0] = m; red[1] = s;
    }
    __syncthreads();
    if (v < VV)
        Ai[((size_t)n * VV + v) * VV + w] += __expf(val - red[0]) / red[1];
}

// AiT[n][w][v] (bf16, 32x32 padded with zeros) from f32 Ai[n][v][w]
__global__ void pack_AiT(const float* __restrict__ Ai, __bf16* __restrict__ AiT) {
    int idx = blockIdx.x * blockDim.x + threadIdx.x;
    if (idx >= NB * 32 * 32) return;
    int n = idx >> 10;
    int r = idx & 1023;
    int w = r >> 5, v = r & 31;
    float val = (w < VV && v < VV) ? Ai[((size_t)n * VV + v) * VV + w] : 0.0f;
    AiT[idx] = (__bf16)val;
}

// =====================================================================
// apply adjacency: out[n,m,w] (+)= sum_v u[n][m][v] * AiT[n][w][v] + bd[m/T]
// M = C*T = 16384 rows, K = 25 padded to 32. grid (M/16, 2, N)
// =====================================================================
__global__ void __launch_bounds__(32)
apply_adj_wmma(const __bf16* __restrict__ u, const __bf16* __restrict__ AiT,
               const float* __restrict__ bd, float* __restrict__ out, int first) {
    int lane = threadIdx.x & 31;
    int hi = lane >> 4, lm = lane & 15;
    int n = blockIdx.z, wT = blockIdx.y, mT = blockIdx.x;
    int wcol = wT * 16 + lm;
    int mrow = mT * 16 + lm;
    const __bf16* ur = u + ((size_t)n * MDIM + mrow) * 32;
    const __bf16* ar = AiT + ((size_t)n * 32 + wcol) * 32;
    v16bf a = load_Afrag(ur, 0, hi);
    v16bf b = load_Bfrag(ar, 0, hi);
    v8f acc = {};
    acc = WMMA_BF16(a, b, acc);
    if (wcol >= VV) return;
#pragma unroll
    for (int r = 0; r < 8; ++r) {
        int m = mT * 16 + (hi << 3) + r;
        int o = m / TT;
        float* p = out + ((size_t)n * MDIM + m) * VV + wcol;
        float val = acc[r] + bd[o];
        *p = first ? val : (*p + val);
    }
}

// =====================================================================
// out = relu( (y - mean)*gamma*rsqrt(var+eps) + beta + x )  in place on y
// =====================================================================
__global__ void bn_relu_res(const float* __restrict__ x,
                            const float* __restrict__ gamma, const float* __restrict__ beta,
                            const float* __restrict__ mean,  const float* __restrict__ var,
                            float* __restrict__ y) {
    size_t idx = (size_t)blockIdx.x * blockDim.x + threadIdx.x;
    size_t total = (size_t)NB * CC * TVC;
    if (idx >= total) return;
    int c = (int)((idx / TVC) % CC);
    float yb = (y[idx] - mean[c]) * gamma[c] * rsqrtf(var[c] + BN_EPS) + beta[c];
    float r = yb + x[idx];
    y[idx] = r > 0.0f ? r : 0.0f;
}

// =====================================================================
extern "C" void kernel_launch(void* const* d_in, const int* in_sizes, int n_in,
                              void* d_out, int out_size, void* d_ws, size_t ws_size,
                              hipStream_t stream) {
    (void)in_sizes; (void)n_in; (void)out_size; (void)ws_size;
    const float* x   = (const float*)d_in[0];
    const float* A   = (const float*)d_in[1];
    const float* PA  = (const float*)d_in[2];
    const float* wa  = (const float*)d_in[3];
    const float* ba  = (const float*)d_in[4];
    const float* wb  = (const float*)d_in[5];
    const float* bb  = (const float*)d_in[6];
    const float* wT1 = (const float*)d_in[7];
    const float* bT1 = (const float*)d_in[8];
    const float* wT2 = (const float*)d_in[9];
    const float* bT2 = (const float*)d_in[10];
    const float* wS1 = (const float*)d_in[11];
    const float* bS1 = (const float*)d_in[12];
    const float* wS2 = (const float*)d_in[13];
    const float* bS2 = (const float*)d_in[14];
    const float* wd  = (const float*)d_in[15];
    const float* bd  = (const float*)d_in[16];
    const float* gm  = (const float*)d_in[17];
    const float* bt  = (const float*)d_in[18];
    const float* mu  = (const float*)d_in[19];
    const float* vr  = (const float*)d_in[20];
    float* out = (float*)d_out;

    // ---- workspace carve (256B aligned chunks) ----
    char* wp = (char*)d_ws;
    auto carve = [&](size_t bytes) -> void* {
        void* r = (void*)wp; wp += (bytes + 255) & ~(size_t)255; return r;
    };
    float*  Aall = (float*) carve(3 * VV * VV * sizeof(float));
    float*  Ai   = (float*) carve((size_t)NB * VV * VV * sizeof(float));
    float*  Sc   = (float*) carve((size_t)NB * VV * VV * sizeof(float));
    __bf16* AiT  = (__bf16*)carve((size_t)NB * 32 * 32 * 2);
    __bf16* xT   = (__bf16*)carve((size_t)NB * TVC * CC * 2);
    __bf16* fT1  = (__bf16*)carve((size_t)NB * 32 * ICT * 2);
    __bf16* fT2  = (__bf16*)carve((size_t)NB * 32 * ICT * 2);
    __bf16* uB   = (__bf16*)carve((size_t)NB * MDIM * 32 * 2);
    __bf16* waB  = (__bf16*)carve((size_t)3 * ICH * CC * 2);
    __bf16* wbB  = (__bf16*)carve((size_t)3 * ICH * CC * 2);
    __bf16* wdB  = (__bf16*)carve((size_t)3 * CC * CC * 2);
    __bf16* wT1B = (__bf16*)carve((size_t)3 * ICH * K9 * 2);
    __bf16* wT2B = (__bf16*)carve((size_t)3 * ICH * K9 * 2);
    __bf16* wS1B = (__bf16*)carve((size_t)3 * ICH * K9 * 2);
    __bf16* wS2B = (__bf16*)carve((size_t)3 * ICH * K9 * 2);

    // ---- one-time preprocessing ----
    {
        size_t tx = (size_t)NB * TVC * CC;
        transpose_x_bf16<<<(unsigned)((tx + 255) / 256), 256, 0, stream>>>(x, xT);
        size_t t1 = (size_t)3 * ICH * CC;
        cvt_bf16<<<(unsigned)((t1 + 255) / 256), 256, 0, stream>>>(wa, waB, t1);
        cvt_bf16<<<(unsigned)((t1 + 255) / 256), 256, 0, stream>>>(wb, wbB, t1);
        size_t td = (size_t)3 * CC * CC;
        cvt_bf16<<<(unsigned)((td + 255) / 256), 256, 0, stream>>>(wd, wdB, td);
        size_t t9 = (size_t)3 * ICH * K9;
        cvt_w9_bf16<<<(unsigned)((t9 + 255) / 256), 256, 0, stream>>>(wT1, wT1B, t9);
        cvt_w9_bf16<<<(unsigned)((t9 + 255) / 256), 256, 0, stream>>>(wT2, wT2B, t9);
        cvt_w9_bf16<<<(unsigned)((t9 + 255) / 256), 256, 0, stream>>>(wS1, wS1B, t9);
        cvt_w9_bf16<<<(unsigned)((t9 + 255) / 256), 256, 0, stream>>>(wS2, wS2B, t9);
        size_t tu = (size_t)NB * MDIM * 32;
        zero_bf16<<<(unsigned)((tu + 255) / 256), 256, 0, stream>>>(uB, tu);
        build_Aall<<<3, dim3(VV, VV), 0, stream>>>(A, PA, Aall);
    }

    const float scale = 1.0f / (float)ICT;
    dim3 gIC(TVC / 32, ICH / 16, NB);   // conv -> inter channels
    dim3 gCC(TVC / 32, CC / 16, NB);    // conv -> full channels
    dim3 gSC(2, 2, NB);                 // attn scores
    dim3 gAD(MDIM / 16, 2, NB);         // apply adjacency

    for (int i = 0; i < 3; ++i) {
        init_Ai<<<(NB * VV * VV + 255) / 256, 256, 0, stream>>>(Aall + i * VV * VV, Ai);

        // ---- branch A1 (1x1 convs) ----
        conv1x1_wmma<<<gIC, 32, 0, stream>>>(xT, waB + (size_t)i * ICH * CC, ba + i * ICH,
                                             fT1, ICH, CC, 0);
        conv1x1_wmma<<<gIC, 32, 0, stream>>>(xT, wbB + (size_t)i * ICH * CC, bb + i * ICH,
                                             fT2, ICH, CC, 0);
        attn_scores_wmma<<<gSC, 32, 0, stream>>>(fT1, fT2, Sc);
        softmax_col_add<<<NB * VV, 32, 0, stream>>>(Sc, Ai, scale);

        // ---- branch AT (temporal 9x1 convs) ----
        conv9_wmma<<<gIC, 32, 0, stream>>>(xT, wT1B + (size_t)i * ICH * K9, bT1 + i * ICH,
                                           fT1, ICH, 0);
        conv9_wmma<<<gIC, 32, 0, stream>>>(xT, wT2B + (size_t)i * ICH * K9, bT2 + i * ICH,
                                           fT2, ICH, 0);
        attn_scores_wmma<<<gSC, 32, 0, stream>>>(fT1, fT2, Sc);
        softmax_col_add<<<NB * VV, 32, 0, stream>>>(Sc, Ai, scale);

        // ---- branch AST (even i: center tap kt=4 slice; odd i: full 9x1) ----
        if ((i & 1) == 0) {
            conv1x1_wmma<<<gIC, 32, 0, stream>>>(xT, wS1B + (size_t)i * ICH * K9 + 4 * CC,
                                                 bS1 + i * ICH, fT1, ICH, K9, 0);
            conv1x1_wmma<<<gIC, 32, 0, stream>>>(xT, wS2B + (size_t)i * ICH * K9 + 4 * CC,
                                                 bS2 + i * ICH, fT2, ICH, K9, 0);
        } else {
            conv9_wmma<<<gIC, 32, 0, stream>>>(xT, wS1B + (size_t)i * ICH * K9,
                                               bS1 + i * ICH, fT1, ICH, 0);
            conv9_wmma<<<gIC, 32, 0, stream>>>(xT, wS2B + (size_t)i * ICH * K9,
                                               bS2 + i * ICH, fT2, ICH, 0);
        }
        attn_scores_wmma<<<gSC, 32, 0, stream>>>(fT1, fT2, Sc);
        softmax_col_add<<<NB * VV, 32, 0, stream>>>(Sc, Ai, scale);

        // ---- u = wd[i] @ x (no bias), then y (+)= u @ Ai + bd[i] ----
        pack_AiT<<<(NB * 32 * 32 + 255) / 256, 256, 0, stream>>>(Ai, AiT);
        conv1x1_wmma<<<gCC, 32, 0, stream>>>(xT, wdB + (size_t)i * CC * CC, nullptr,
                                             uB, CC, CC, 1);
        apply_adj_wmma<<<gAD, 32, 0, stream>>>(uB, AiT, bd + i * CC, out, i == 0 ? 1 : 0);
    }

    size_t total = (size_t)NB * CC * TVC;
    bn_relu_res<<<(unsigned)((total + 255) / 256), 256, 0, stream>>>(x, gm, bt, mu, vr, out);
}
